// GCN_54185307406447
// MI455X (gfx1250) — compile-verified, matrix operand order checked by reading, not run on hardware
//
#include <hip/hip_runtime.h>
#include <hip/hip_bf16.h>

#define F_DIM   128
#define B_BATCH 8

typedef __attribute__((ext_vector_type(16))) __bf16 v16bf;
typedef __attribute__((ext_vector_type(8)))  float  v8f;

union BF16x16 { unsigned short s[16]; uint4 q[2]; v16bf v; };
union Half8   { unsigned short s[8];  uint4 q; };

__device__ __forceinline__ unsigned short bf16_bits(float f) {
  __bf16 b = (__bf16)f;
  return __builtin_bit_cast(unsigned short, b);
}

__device__ __forceinline__ void atomic_add_f32(float* p, float v) {
  unsafeAtomicAdd(p, v);                      // global_atomic_add_f32
}

// ---------------- degree / symmetric norm ----------------
__global__ void deg_init_kernel(float* deg, int BN) {
  int i = blockIdx.x * blockDim.x + threadIdx.x;
  if (i < BN) deg[i] = 1.0f;                  // self-loop on every node
}

__global__ void deg_accum_kernel(const int* __restrict__ dst, float* deg, int E) {
  int e = blockIdx.x * blockDim.x + threadIdx.x;
  if (e < E) atomic_add_f32(&deg[dst[e]], (float)B_BATCH);  // B tiled copies
}

__global__ void deg_to_dinv_kernel(float* deg, int BN) {
  int i = blockIdx.x * blockDim.x + threadIdx.x;
  if (i < BN) deg[i] = rsqrtf(deg[i]);        // deg >= 1 always
}

// ---------------- W -> bf16 B-fragments (once, 32 KB, L2-resident) ----------------
// wf[tile][kk][lane] = 16 bf16 = exact v16bf B operand for that wave/kstep.
// B fragment (ISA 7.12.2): lane L -> col n=L&15, khalf=(L>>4)*8;
// element e -> K = kk*32 + khalf + (e<8 ? e : 8+e); value = W[tile*16+n][K].
__global__ void wconv_kernel(const float* __restrict__ W,
                             unsigned short* __restrict__ wf) {
  int t = blockIdx.x * blockDim.x + threadIdx.x;   // 2048 threads total
  int half = t & 1;
  int frag = t >> 1;            // 0..1023 = [tile][kk][lane]
  int lane = frag & 31;
  int kk   = (frag >> 5) & 3;
  int tile = frag >> 7;
  int n     = lane & 15;
  int khalf = (lane >> 4) << 3;
  int kbase = kk * 32 + khalf + (half ? 16 : 0);
  const float* wrow = W + (size_t)(tile * 16 + n) * F_DIM + kbase;
  Half8 h8;
#pragma unroll
  for (int e = 0; e < 8; ++e) h8.s[e] = bf16_bits(wrow[e]);
  *(uint4*)(wf + (size_t)frag * 16 + half * 8) = h8.q;
}

// ---------------- h = x @ W^T  (bf16 WMMA) + fused self-loop/bias epilogue ----
// Block = 256 threads = 8 waves, owns 16 rows. Stage A fragments in LDS once;
// wave w computes cols [16w,16w+16) with 4 WMMAs (B streamed from wf).
// Epilogue writes h AND out = dinv^2 * h + b, saving a full pass over h.
__global__ void __launch_bounds__(256)
gemm_wmma_kernel(const float* __restrict__ x,
                 const unsigned short* __restrict__ wf,
                 const float* __restrict__ dinv,
                 const float* __restrict__ bias,
                 float* __restrict__ h,
                 float* __restrict__ out) {
  __shared__ __align__(16) unsigned short ldsA[4 * 32 * 16];   // 4 KB

  const int tid  = threadIdx.x;
  const int wave = tid >> 5;
  const int lane = tid & 31;
  const int row_base = blockIdx.x * 16;

  // --- stage A: each thread converts 8 contiguous floats (half a fragment) ---
  {
    int half  = tid & 1;
    int frag  = tid >> 1;        // 0..127 = [kk][lane]
    int flane = frag & 31;
    int kk    = frag >> 5;
    int row   = flane & 15;
    int khalf = (flane >> 4) << 3;
    int kbase = kk * 32 + khalf + (half ? 16 : 0);
    const float* xr = x + (size_t)(row_base + row) * F_DIM + kbase;
    Half8 h8;
#pragma unroll
    for (int e = 0; e < 8; ++e) h8.s[e] = bf16_bits(xr[e]);
    *(uint4*)(ldsA + frag * 16 + half * 8) = h8.q;
  }
  __syncthreads();

  const unsigned short* wtile = wf + (size_t)wave * (4 * 32 * 16);

  v8f acc = {};
#pragma unroll
  for (int kk = 0; kk < 4; ++kk) {
    BF16x16 a, b;
    a.q[0] = *(const uint4*)(ldsA  + (kk * 32 + lane) * 16);
    a.q[1] = *(const uint4*)(ldsA  + (kk * 32 + lane) * 16 + 8);
    b.q[0] = *(const uint4*)(wtile + (kk * 32 + lane) * 16);
    b.q[1] = *(const uint4*)(wtile + (kk * 32 + lane) * 16 + 8);
    acc = __builtin_amdgcn_wmma_f32_16x16x32_bf16(false, a.v, false, b.v,
                                                  (short)0, acc, false, false);
  }

  // D layout: lanes 0-15: N=lane, M=r; lanes 16-31: N=lane-16, M=r+8.
  const int n   = lane & 15;
  const int mo  = (lane >> 4) << 3;
  const int col = (wave << 4) + n;
  const float bv = bias[col];
#pragma unroll
  for (int r = 0; r < 8; ++r) {
    int row = row_base + mo + r;
    float di = dinv[row];                      // 16 hot values per block
    size_t off = (size_t)row * F_DIM + col;
    h[off]   = acc[r];
    out[off] = acc[r] * (di * di) + bv;        // self-loop + bias, fused
  }
}

// ---------------- edge gather + scatter-add (one wave per edge) ----------------
__global__ void edge_scatter_kernel(const float* __restrict__ h,
                                    const float* __restrict__ dinv,
                                    const int* __restrict__ src,
                                    const int* __restrict__ dst,
                                    float* out, int E) {
  int t = blockIdx.x * blockDim.x + threadIdx.x;
  int e = t >> 5;
  if (e >= E) return;
  int lane = t & 31;
  int s = src[e], d = dst[e];
  float nrm = dinv[s] * dinv[d] * (float)B_BATCH;   // fold B tiled copies
  float4 v = *(const float4*)(h + (size_t)s * F_DIM + (lane << 2));
  float* o = out + (size_t)d * F_DIM + (lane << 2);
  atomic_add_f32(o + 0, v.x * nrm);
  atomic_add_f32(o + 1, v.y * nrm);
  atomic_add_f32(o + 2, v.z * nrm);
  atomic_add_f32(o + 3, v.w * nrm);
}

extern "C" void kernel_launch(void* const* d_in, const int* in_sizes, int n_in,
                              void* d_out, int out_size, void* d_ws, size_t ws_size,
                              hipStream_t stream) {
  const float* data = (const float*)d_in[0];  // [B, N, 128] f32
  const int*   ei   = (const int*)d_in[1];    // [2, E] i32
  const float* W    = (const float*)d_in[2];  // [128, 128] f32
  const float* bia  = (const float*)d_in[3];  // [128] f32
  float* out = (float*)d_out;

  const int BN = in_sizes[0] / F_DIM;         // 80000
  const int E  = in_sizes[1] / 2;             // 160000
  const int* src = ei;
  const int* dst = ei + E;

  // workspace: h [BN*128 f32] | dinv [BN f32] | wf [8*4*32*16 bf16]
  char*  ws   = (char*)d_ws;
  float* h    = (float*)ws;
  float* dinv = (float*)(ws + (size_t)BN * F_DIM * sizeof(float));
  unsigned short* wfrag =
      (unsigned short*)(ws + (size_t)BN * F_DIM * sizeof(float)
                           + (size_t)((BN + 63) & ~63) * sizeof(float));

  deg_init_kernel   <<<(BN + 255) / 256, 256, 0, stream>>>(dinv, BN);
  deg_accum_kernel  <<<(E  + 255) / 256, 256, 0, stream>>>(dst, dinv, E);
  deg_to_dinv_kernel<<<(BN + 255) / 256, 256, 0, stream>>>(dinv, BN);

  wconv_kernel<<<8, 256, 0, stream>>>(W, wfrag);
  gemm_wmma_kernel<<<BN / 16, 256, 0, stream>>>(data, wfrag, dinv, bia, h, out);

  long long tEdge = (long long)E * 32;        // one wave per edge
  edge_scatter_kernel<<<(int)((tEdge + 255) / 256), 256, 0, stream>>>(h, dinv, src, dst, out, E);
}
